// FeedForwardQuantum_65481071396002
// MI455X (gfx1250) — matrix operand order repaired and checked by reading, not compiled, not run on hardware
//
#include <hip/hip_runtime.h>
#include <hip/hip_bf16.h>
#include <math.h>

typedef __attribute__((ext_vector_type(16))) _Float16 v16h;
typedef __attribute__((ext_vector_type(8)))  _Float16 v8h;
typedef __attribute__((ext_vector_type(8)))  float    v8f;
typedef __attribute__((ext_vector_type(4)))  unsigned int uint32x4;
typedef __attribute__((ext_vector_type(8)))  int          int32x8;
typedef __attribute__((ext_vector_type(4)))  int          int32x4;

#define EMBED  1024
#define FFN    4096
#define NQ     10
#define TILE_M 128
#define TILE_N 128
#define KC     64
#define LDA    72   // padded LDS stride in halfs (144 B = 128 B data + 16 B pad)

// ---------------------------------------------------------------------------
// Kernel 0: one-time w2 f32 -> f16 conversion into workspace (read 128x later)
// ---------------------------------------------------------------------------
__global__ __launch_bounds__(256)
void w2cvt_kernel(const float* __restrict__ w2, _Float16* __restrict__ w2h, int n4)
{
    int t = blockIdx.x * 256 + threadIdx.x;     // one thread per 4 elements
    if (t >= n4) return;
    const float4 v = *(const float4*)(&w2[(size_t)t * 4]);
    union { _Float16 f[4]; uint2 u; } pk;
    pk.f[0] = (_Float16)v.x; pk.f[1] = (_Float16)v.y;
    pk.f[2] = (_Float16)v.z; pk.f[3] = (_Float16)v.w;
    *(uint2*)(&w2h[(size_t)t * 4]) = pk.u;
}

// ---------------------------------------------------------------------------
// Kernel 1: q_out[tok][q] = cos(x[tok][q]) * cos(theta[q])   (tok<16384, q<10)
// ---------------------------------------------------------------------------
__global__ __launch_bounds__(256)
void qenc_kernel(const float* __restrict__ x, const float* __restrict__ theta,
                 float* __restrict__ qout, int ntok)
{
    int t = blockIdx.x * 256 + threadIdx.x;
    if (t >= ntok * NQ) return;
    int tok = t / NQ;
    int q   = t - tok * NQ;
    qout[t] = cosf(x[(size_t)tok * EMBED + q]) * cosf(theta[q]);
}

// ---------------------------------------------------------------------------
// Kernel 2: fused  h = relu(qout @ w1^T + b1)  ;  y = h @ w2^T + b2
// 256 threads = 8 wave32s, 128x128 output tile, K chunked by 64.
// B chunk (w2h, f16) is DMA'd by the Tensor Data Mover into padded LDS
// (TENSORcnt), overlapping with the f32 on-the-fly A (h) tile computation.
// WMMA: v_wmma_f32_16x16x32_f16 on fragments per CDNA5 ISA 7.12.2 layouts.
// ---------------------------------------------------------------------------
__global__ __launch_bounds__(256)
void ffnq_gemm(const float* __restrict__ qout,      // [16384][10]
               const float* __restrict__ w1,        // [4096][10]
               const float* __restrict__ b1,        // [4096]
               const _Float16* __restrict__ w2h,    // [1024][4096] f16
               const float* __restrict__ b2,        // [1024]
               float* __restrict__ y)               // [16384][1024]
{
    __shared__ __align__(16) _Float16 Al[TILE_M * LDA];  // A tile [m][k]
    __shared__ __align__(16) _Float16 Bl[TILE_N * LDA];  // B tile [n][k]
    __shared__ float qo [TILE_M * NQ];
    __shared__ float w1l[KC * NQ];
    __shared__ float b1l[KC];

    const int tid  = threadIdx.x;
    const int lane = tid & 31;
    const int wave = tid >> 5;
    const int wm   = wave & 3;     // 4 bands of 32 rows
    const int wn   = wave >> 2;    // 2 bands of 64 cols
    const int mtile = blockIdx.y * TILE_M;
    const int ntile = blockIdx.x * TILE_N;

    // Stage qout tile once (1280 f32)
    for (int i = tid; i < TILE_M * NQ; i += 256) qo[i] = qout[(size_t)mtile * NQ + i];
    __syncthreads();

    // Fixed per-thread h-compute assignment: row hm, k in [hk0, hk0+32)
    const int hm  = tid >> 1;
    const int hk0 = (tid & 1) * 32;
    float qreg[NQ];
#pragma unroll
    for (int j = 0; j < NQ; ++j) qreg[j] = qo[hm * NQ + j];

    v8f acc[2][4];
    const v8f vzero = {0.f, 0.f, 0.f, 0.f, 0.f, 0.f, 0.f, 0.f};
#pragma unroll
    for (int im = 0; im < 2; ++im)
#pragma unroll
        for (int in = 0; in < 4; ++in) acc[im][in] = vzero;

    // WMMA fragment lane geometry (ISA 7.12.2, wave32)
    const int r   = lane & 15;      // row (A) / col (B) / col (C,D)
    const int hi  = lane >> 4;      // half-wave select
    const int kbA = hi * 8;         // A: K-groups {kb..kb+7, kb+16..kb+23}
    const int kbB = hi * 8;         // B: K-groups {kb..kb+7, kb+16..kb+23}

#if __has_builtin(__builtin_amdgcn_tensor_load_to_lds)
    // Tensor DMA descriptor pieces that do not change across K chunks.
    const unsigned lds_b = (unsigned)(size_t)(void*)Bl;
    // group1: data_size=2B, pad_enable, pad_interval=32 DW (code 4),
    // pad_amount=4 DW (code 3); tensor_dim0=4096, tensor_dim1=1024,
    // tile_dim0=64, tile_dim1=128, tile_dim2=0; tensor_dim0_stride=4096.
    int32x8 g1;
    g1[0] = (int)((1u << 16) | (1u << 20) | (4u << 22) | (3u << 25));
    g1[1] = (int)(((unsigned)FFN & 0xFFFFu) << 16);                  // dim0 lo16
    g1[2] = (int)(((unsigned)FFN >> 16) | ((1024u & 0xFFFFu) << 16));// dim0 hi | dim1 lo
    g1[3] = (int)((1024u >> 16) | (64u << 16));                      // dim1 hi | tile_dim0
    g1[4] = (int)(128u);                                             // tile_dim1 | tile_dim2=0
    g1[5] = (int)FFN;                                                // dim0_stride lo32
    g1[6] = 0;
    g1[7] = 0;
    const int32x4 g2 = {0, 0, 0, 0};
    const int32x4 g3 = {0, 0, 0, 0};
    const int32x8 g4 = {0, 0, 0, 0, 0, 0, 0, 0};   // 6-arg toolchain extra group
#endif

    for (int kc = 0; kc < FFN; kc += KC) {
        __syncthreads();  // previous iteration's WMMA reads done before LDS overwrite

#if __has_builtin(__builtin_amdgcn_tensor_load_to_lds)
        if (wave == 0) {
            // group0: count=1 (user), lds_addr, 57-bit global_addr, type=2
            const unsigned long long ga =
                (unsigned long long)(size_t)(const void*)&w2h[(size_t)ntile * FFN + kc];
            uint32x4 g0;
            g0[0] = 1u;
            g0[1] = lds_b;
            g0[2] = (unsigned)(ga & 0xFFFFFFFFu);
            g0[3] = (unsigned)((ga >> 32) & 0x01FFFFFFu) | (2u << 30);
            __builtin_amdgcn_tensor_load_to_lds(g0, g1, g2, g3, g4, 0);
        }
#else
        // Fallback: manual 16B copies of the f16 B chunk into padded LDS
#pragma unroll
        for (int it = 0; it < 4; ++it) {
            int idx = tid + it * 256;           // 0..1023 16B chunks
            int n   = idx >> 3;
            int c   = idx & 7;
            *(uint4*)(&Bl[n * LDA + c * 8]) =
                *(const uint4*)(&w2h[(size_t)(ntile + n) * FFN + kc + c * 8]);
        }
#endif

        // Stage w1/b1 chunk (640 + 64 f32) — overlaps with the tensor DMA
        for (int i = tid; i < KC * NQ; i += 256) w1l[i] = w1[(size_t)kc * NQ + i];
        if (tid < KC) b1l[tid] = b1[kc + tid];
        __syncthreads();  // w1l/b1l visible

        // Compute A chunk: h[m][k] = relu(dot10(qout[m], w1[k]) + b1[k]) -> f16 LDS
        for (int e = 0; e < 32; ++e) {
            int k = hk0 + e;
            float s = b1l[k];
#pragma unroll
            for (int j = 0; j < NQ; ++j) s += qreg[j] * w1l[k * NQ + j];
            s = s > 0.f ? s : 0.f;
            Al[hm * LDA + k] = (_Float16)s;
        }

#if __has_builtin(__builtin_amdgcn_tensor_load_to_lds)
        if (wave == 0) __builtin_amdgcn_s_wait_tensorcnt(0);  // B chunk landed
#endif
        __syncthreads();  // A and B tiles visible to all waves

        // Tensor-core stage: 2 K-steps of 32, 2x4 fragments per wave
#pragma unroll
        for (int ks = 0; ks < KC; ks += 32) {
            v16h a[2], b[4];
#pragma unroll
            for (int im = 0; im < 2; ++im) {
                const _Float16* ap = &Al[(wm * 32 + im * 16 + r) * LDA + ks + kbA];
                union { v16h v; v8h h[2]; } u;
                u.h[0] = *(const v8h*)(ap);
                u.h[1] = *(const v8h*)(ap + 16);
                a[im] = u.v;
            }
#pragma unroll
            for (int in = 0; in < 4; ++in) {
                const _Float16* bp = &Bl[(wn * 64 + in * 16 + r) * LDA + ks + kbB];
                union { v16h v; v8h h[2]; } u;
                u.h[0] = *(const v8h*)(bp);
                u.h[1] = *(const v8h*)(bp + 16);
                b[in] = u.v;
            }
#pragma unroll
            for (int im = 0; im < 2; ++im)
#pragma unroll
                for (int in = 0; in < 4; ++in)
                    acc[im][in] = __builtin_amdgcn_wmma_f32_16x16x32_f16(
                        false, a[im], false, b[in], (short)0, acc[im][in], false, false);
        }
    }

    // Epilogue: + b2, store f32. D layout: VGPR i -> row 8*hi+i, col = lane&15.
    const int mb = mtile + wm * 32;
    const int nb = ntile + wn * 64;
#pragma unroll
    for (int in = 0; in < 4; ++in) {
        const int col = nb + in * 16 + r;
        const float bias = b2[col];
#pragma unroll
        for (int im = 0; im < 2; ++im) {
#pragma unroll
            for (int i = 0; i < 8; ++i) {
                const int row = mb + im * 16 + hi * 8 + i;
                y[(size_t)row * EMBED + col] = acc[im][in][i] + bias;
            }
        }
    }
}

// ---------------------------------------------------------------------------
extern "C" void kernel_launch(void* const* d_in, const int* in_sizes, int n_in,
                              void* d_out, int out_size, void* d_ws, size_t ws_size,
                              hipStream_t stream)
{
    const float* x     = (const float*)d_in[0];  // [B,S,1024]
    const float* theta = (const float*)d_in[1];  // [10]
    const float* w1    = (const float*)d_in[2];  // [4096,10]
    const float* b1    = (const float*)d_in[3];  // [4096]
    const float* w2    = (const float*)d_in[4];  // [1024,4096]
    const float* b2    = (const float*)d_in[5];  // [1024]
    float* y = (float*)d_out;                    // [B,S,1024]

    float*     qout = (float*)d_ws;                            // 640 KB
    _Float16*  w2h  = (_Float16*)((char*)d_ws + (1u << 20));   // 8 MB

    const int ntok = in_sizes[0] / EMBED;        // 16384

    w2cvt_kernel<<<(EMBED * FFN / 4 + 255) / 256, 256, 0, stream>>>(w2, w2h, EMBED * FFN / 4);
    qenc_kernel<<<(ntok * NQ + 255) / 256, 256, 0, stream>>>(x, theta, qout, ntok);

    dim3 grid(EMBED / TILE_N, ntok / TILE_M);    // (8, 128)
    ffnq_gemm<<<grid, 256, 0, stream>>>(qout, w1, b1, w2h, b2, y);
}
